// FM_81140522156059
// MI455X (gfx1250) — matrix operand order repaired and checked by reading, not compiled, not run on hardware
//
#include <hip/hip_runtime.h>
#include <hip/hip_bf16.h>
#include <math.h>

// Problem sizes (fixed by the reference)
#define BB 4096
#define FF 8192
#define KK 64
#define FSPLIT 16              // F-chunks across gridDim.y
#define FCHUNK (FF / FSPLIT)   // 512 columns per block
#define CHC 256                // columns staged into LDS per stage (2 stages)
#define LPITCH 264             // bf16 elements per LDS row (256 + 8 pad -> bank stride 4)

typedef __attribute__((ext_vector_type(16))) __bf16 v16bf;
typedef __attribute__((ext_vector_type(8)))  __bf16 v8bf;
typedef __attribute__((ext_vector_type(8)))  float  v8f;
typedef __attribute__((ext_vector_type(4)))  float  f32x4;

// ---------------------------------------------------------------------------
// Kernel 0: zero the accumulator region of workspace (XV, linear, diag).
// Must run every launch: graph replays do not re-initialize d_ws.
// ---------------------------------------------------------------------------
__global__ void fm_zero(float* __restrict__ p, int n) {
    int i = blockIdx.x * blockDim.x + threadIdx.x;
    if (i < n) p[i] = 0.0f;
}

// ---------------------------------------------------------------------------
// Kernel 1: prep V.  Vt[k][i] = bf16(V[i][k])  (transposed, bf16, L2-resident)
//           vnorm2[i] = sum_k V[i][k]^2
// One wave (32 lanes) per feature row i; each lane covers k and k+32.
// ---------------------------------------------------------------------------
__global__ void fm_prep(const float* __restrict__ V,
                        __bf16* __restrict__ vt,
                        float* __restrict__ vnorm2) {
    const int lane = threadIdx.x & 31;
    const int wid  = threadIdx.x >> 5;
    const int i    = blockIdx.x * 8 + wid;          // feature index, 8 rows/block
    float v0 = V[(size_t)i * KK + lane];
    float v1 = V[(size_t)i * KK + lane + 32];
    vt[(size_t)lane        * FF + i] = (__bf16)v0;
    vt[(size_t)(lane + 32) * FF + i] = (__bf16)v1;
    float s = v0 * v0 + v1 * v1;
    #pragma unroll
    for (int off = 16; off; off >>= 1) s += __shfl_down(s, off, 32);
    if (lane == 0) vnorm2[i] = s;
}

// ---------------------------------------------------------------------------
// Kernel 2: fused main pass over x (read once from HBM).
//   - XV += X_tile @ V_chunk via v_wmma_f32_16x16x32_bf16 (atomic f32 accum)
//   - linear[b] += x[b,:chunk] . W[:chunk]
//   - diag      += sum x^2 * vnorm2  (scalar)
// Grid: (B/128, FSPLIT), 256 threads = 8 waves; wave owns 16 rows x 64 cols.
// The block's V^T chunk is staged into LDS with GLOBAL_LOAD_ASYNC_TO_LDS_B128
// (ASYNCcnt-tracked) in two 32KB stages; B fragments are then ds_load'ed.
// All 4 B fragments are loaded before the 4 WMMAs so the ds_loads clause up
// behind one s_wait_dscnt and the WMMAs issue back-to-back (no D->A/B hazard:
// shared A, disjoint accumulators).
//
// NOTE: the LDS destination offset passed to the async asm is derived from
// &vsh (ptrtoint) so the LDS object escapes into the asm. Without that, LLVM
// proves vsh is never written and folds all fragment loads to undef.
//
// Fragment layout assumptions (CDNA5 ISA 7.12.2, wave32):
//   A 16x32 bf16: lane l holds row m=l&15; K-runs [8h..8h+7] and [16+8h..+7],
//                 h = l>>4, packed pairs per VGPR (elements in that order).
//   B 32x16 bf16: lane l holds col n=l&15, K-runs [8h..8h+7] and [16+8h..+7] (same as A).
//   C/D f32 16x16: VGPR r holds row (r + 8*(l>=16)), col n=l&15.
// ---------------------------------------------------------------------------
__global__ void __launch_bounds__(256)
fm_main(const float* __restrict__ x, const float* __restrict__ W,
        const __bf16* __restrict__ vt, const float* __restrict__ vnorm2,
        float* __restrict__ XV, float* __restrict__ lin,
        float* __restrict__ diag) {
    __shared__ __bf16 vsh[64 * LPITCH];   // 33,792 bytes

    const int lane = threadIdx.x & 31;
    const int wid  = threadIdx.x >> 5;
    const int m    = lane & 15;
    const int h    = lane >> 4;
    const int row0 = blockIdx.x * 128 + wid * 16;
    const int f0   = blockIdx.y * FCHUNK;

    v8f acc0 = {}, acc1 = {}, acc2 = {}, acc3 = {};
    float lpart = 0.0f, dpart = 0.0f;

    const float* xrow = x + (size_t)(row0 + m) * FF;

    for (int s = 0; s < FCHUNK / CHC; ++s) {
        const int fs = f0 + s * CHC;          // global column base of this stage

        __syncthreads();                      // previous-stage readers done
        // ---- async stage: 64 rows x 256 bf16 = 32KB, 512B per wave-instr ----
        #pragma unroll
        for (int i = 0; i < 8; ++i) {
            const int c   = (int)threadIdx.x + 256 * i;   // 16B-chunk id
            const int row = c >> 5;                       // 32 chunks per row
            const int col = (c & 31) * 8;                 // bf16 element col
            const __bf16* g = vt + (size_t)row * FF + fs + col;
            // LDS offset derived from the shared object itself (escape!):
            // low 32 bits of the generic LDS aperture address == workgroup-
            // relative LDS byte offset (aperture base has zero low bits).
            const unsigned loff = (unsigned)(size_t)(&vsh[row * LPITCH + col]);
            asm volatile("global_load_async_to_lds_b128 %0, %1, off"
                         :: "v"(loff), "v"(g) : "memory");
        }
#if defined(__has_builtin) && __has_builtin(__builtin_amdgcn_s_wait_asynccnt)
        __builtin_amdgcn_s_wait_asynccnt(0);
#else
        asm volatile("s_wait_asynccnt 0x0" ::: "memory");
#endif
        __syncthreads();                      // stage visible to all waves

        for (int kk = 0; kk < CHC; kk += 32) {
            const int kb = fs + kk;           // global K base (x/W/vnorm2)
            const int o0 = kb + 8 * h;        // first  8-run
            const int o1 = kb + 16 + 8 * h;   // second 8-run

            // prefetch next K-step of the HBM-critical x stream
            __builtin_prefetch(xrow + kb + 256, 0, 3);

            // --- all 4 B fragments from LDS first (clause + single wait) ---
            const int bo = kk + 8 * h;        // LDS-local column
            const __bf16* bp0 = &vsh[(0 * 16 + m) * LPITCH + bo];
            const __bf16* bp1 = &vsh[(1 * 16 + m) * LPITCH + bo];
            const __bf16* bp2 = &vsh[(2 * 16 + m) * LPITCH + bo];
            const __bf16* bp3 = &vsh[(3 * 16 + m) * LPITCH + bo];
            v8bf b0a = *(const v8bf*)bp0, b0b = *(const v8bf*)(bp0 + 16);
            v8bf b1a = *(const v8bf*)bp1, b1b = *(const v8bf*)(bp1 + 16);
            v8bf b2a = *(const v8bf*)bp2, b2b = *(const v8bf*)(bp2 + 16);
            v8bf b3a = *(const v8bf*)bp3, b3b = *(const v8bf*)(bp3 + 16);
            v16bf bf0, bf1, bf2, bf3;
            #pragma unroll
            for (int i = 0; i < 8; ++i) {
                bf0[i] = b0a[i]; bf0[8 + i] = b0b[i];
                bf1[i] = b1a[i]; bf1[8 + i] = b1b[i];
                bf2[i] = b2a[i]; bf2[8 + i] = b2b[i];
                bf3[i] = b3a[i]; bf3[8 + i] = b3b[i];
            }

            // --- A fragment: 16 fp32 of x, convert to bf16 ---
            f32x4 a0 = *(const f32x4*)(xrow + o0);
            f32x4 a1 = *(const f32x4*)(xrow + o0 + 4);
            f32x4 a2 = *(const f32x4*)(xrow + o1);
            f32x4 a3 = *(const f32x4*)(xrow + o1 + 4);
            v16bf af;
            #pragma unroll
            for (int i = 0; i < 4; ++i) {
                af[i]      = (__bf16)a0[i];
                af[4 + i]  = (__bf16)a1[i];
                af[8 + i]  = (__bf16)a2[i];
                af[12 + i] = (__bf16)a3[i];
            }

            // --- 4 WMMAs back-to-back ---
            acc0 = __builtin_amdgcn_wmma_f32_16x16x32_bf16(false, af, false, bf0,
                                                           (short)0, acc0, false, false);
            acc1 = __builtin_amdgcn_wmma_f32_16x16x32_bf16(false, af, false, bf1,
                                                           (short)0, acc1, false, false);
            acc2 = __builtin_amdgcn_wmma_f32_16x16x32_bf16(false, af, false, bf2,
                                                           (short)0, acc2, false, false);
            acc3 = __builtin_amdgcn_wmma_f32_16x16x32_bf16(false, af, false, bf3,
                                                           (short)0, acc3, false, false);

            // --- fused linear + diag partials (W, vnorm2 are L2-resident),
            //     co-executes with the matrix pipe ---
            f32x4 w0 = *(const f32x4*)(W + o0);
            f32x4 w1 = *(const f32x4*)(W + o0 + 4);
            f32x4 w2 = *(const f32x4*)(W + o1);
            f32x4 w3 = *(const f32x4*)(W + o1 + 4);
            f32x4 n0 = *(const f32x4*)(vnorm2 + o0);
            f32x4 n1 = *(const f32x4*)(vnorm2 + o0 + 4);
            f32x4 n2 = *(const f32x4*)(vnorm2 + o1);
            f32x4 n3 = *(const f32x4*)(vnorm2 + o1 + 4);
            #pragma unroll
            for (int i = 0; i < 4; ++i) {
                lpart += a0[i]*w0[i] + a1[i]*w1[i] + a2[i]*w2[i] + a3[i]*w3[i];
                dpart += a0[i]*a0[i]*n0[i] + a1[i]*a1[i]*n1[i]
                       + a2[i]*a2[i]*n2[i] + a3[i]*a3[i]*n3[i];
            }
        }
    }

    // --- accumulate partial XV tiles (F is split across gridDim.y) ---
    const int rowA = row0 + ((lane < 16) ? 0 : 8);
    #pragma unroll
    for (int r = 0; r < 8; ++r) {
        atomicAdd(&XV[(size_t)(rowA + r) * KK +  0 + m], acc0[r]);
        atomicAdd(&XV[(size_t)(rowA + r) * KK + 16 + m], acc1[r]);
        atomicAdd(&XV[(size_t)(rowA + r) * KK + 32 + m], acc2[r]);
        atomicAdd(&XV[(size_t)(rowA + r) * KK + 48 + m], acc3[r]);
    }

    // --- linear: lanes m and m+16 hold the two K-halves of row m ---
    lpart += __shfl_xor(lpart, 16, 32);
    if (lane < 16) atomicAdd(&lin[row0 + m], lpart);

    // --- diag scalar: full-wave reduce, one atomic per wave ---
    #pragma unroll
    for (int off = 16; off; off >>= 1) dpart += __shfl_down(dpart, off, 32);
    if (lane == 0) atomicAdd(diag, dpart);
}

// ---------------------------------------------------------------------------
// Kernel 3: finalize.  cross = 0.5*(||XV||_F^2 - diag); out = sigmoid(lin+b+cross)
// Single block; the reductions are over only 256K + 4K floats (L2-resident).
// ---------------------------------------------------------------------------
__global__ void fm_finalize(const float* __restrict__ XV,
                            const float* __restrict__ lin,
                            const float* __restrict__ diag,
                            const float* __restrict__ bias,
                            float* __restrict__ out) {
    __shared__ float red[32];
    __shared__ float crossS;
    float s = 0.0f;
    for (int i = threadIdx.x; i < BB * KK; i += blockDim.x) {
        float v = XV[i];
        s += v * v;
    }
    #pragma unroll
    for (int off = 16; off; off >>= 1) s += __shfl_down(s, off, 32);
    const int lane = threadIdx.x & 31, wid = threadIdx.x >> 5;
    if (lane == 0) red[wid] = s;
    __syncthreads();
    if (threadIdx.x == 0) {
        float t = 0.0f;
        for (int i = 0; i < (int)(blockDim.x >> 5); ++i) t += red[i];
        crossS = 0.5f * (t - *diag) + *bias;
    }
    __syncthreads();
    for (int i = threadIdx.x; i < BB; i += blockDim.x) {
        float z = lin[i] + crossS;
        out[i] = 1.0f / (1.0f + expf(-z));
    }
}

// ---------------------------------------------------------------------------
// Host-side launcher
// ---------------------------------------------------------------------------
extern "C" void kernel_launch(void* const* d_in, const int* in_sizes, int n_in,
                              void* d_out, int out_size, void* d_ws, size_t ws_size,
                              hipStream_t stream) {
    const float* x    = (const float*)d_in[0];   // [B, F]
    const float* W    = (const float*)d_in[1];   // [1, F]
    const float* bias = (const float*)d_in[2];   // [1]
    const float* V    = (const float*)d_in[3];   // [F, K]
    float* out = (float*)d_out;                  // [B]

    // Workspace layout
    char* ws = (char*)d_ws;
    __bf16* vt     = (__bf16*)ws;                          // F*K bf16  (1 MB)
    float*  vnorm2 = (float*)(ws + (size_t)FF * KK * 2);   // F floats
    float*  XV     = vnorm2 + FF;                          // B*K floats (1 MB)
    float*  lin    = XV + (size_t)BB * KK;                 // B floats
    float*  diag   = lin + BB;                             // 1 float

    // 0) zero XV, lin, diag (contiguous region)
    const int nzero = BB * KK + BB + 1;
    fm_zero<<<(nzero + 255) / 256, 256, 0, stream>>>(XV, nzero);

    // 1) V -> Vt(bf16) + vnorm2
    fm_prep<<<FF / 8, 256, 0, stream>>>(V, vt, vnorm2);

    // 2) fused single pass over x with WMMA + async LDS staging of V
    dim3 grid(BB / 128, FSPLIT);
    fm_main<<<grid, 256, 0, stream>>>(x, W, vt, vnorm2, XV, lin, diag);

    // 3) finalize
    fm_finalize<<<1, 1024, 0, stream>>>(XV, lin, diag, bias, out);
}
